// CCMNet_16106127360571
// MI455X (gfx1250) — matrix operand order repaired and checked by reading, not compiled, not run on hardware
//
#include <hip/hip_runtime.h>
#include <math.h>

// Problem constants (match reference)
#define B_    4
#define CIN   64
#define COUT  64
#define NPTS  4096
#define RG    32
#define RV3   (RG * RG * RG)   // 32768
#define KNN   16
#define RAD2  0.04f            // RADIUS^2
#define BNEPS 1e-4f

typedef __attribute__((ext_vector_type(16))) _Float16 v16h;
typedef __attribute__((ext_vector_type(8)))  float    v8f;

// ---- WMMA fragment index helpers (CDNA5 ISA 7.12.2, wave32) -----------------
// A (16x32 f16), this code's k order (not the ISA table's; B matches): lanes 0-15 -> M=lane, K{0..15}; lanes 16-31 -> K{16..31}
__device__ __forceinline__ int a_kidx(int lane, int i) {
  int base = (lane >> 4) * 16;
  return base + i;
}

// ---- small utility kernels --------------------------------------------------
__global__ void k_zero(float* __restrict__ p, long n) {
  long i = (long)blockIdx.x * blockDim.x + threadIdx.x;
  if (i < n) p[i] = 0.0f;
}

__global__ void k_copy(const float* __restrict__ s, float* __restrict__ d, int n) {
  int i = blockIdx.x * blockDim.x + threadIdx.x;
  if (i < n) d[i] = s[i];
}

// ---- weight repack: f32 -> f16 in WMMA A-fragment order ---------------------
// GEMM weights w[co*K + k] -> dst[frag][lane][i], frag = chunk*4 + coTile
__global__ void k_pack_a_gemm(const float* __restrict__ w, _Float16* __restrict__ dst, int K) {
  int nthr = (K >> 5) * 4 * 32;
  int tid = blockIdx.x * blockDim.x + threadIdx.x;
  if (tid >= nthr) return;
  int lane  = tid & 31;
  int f     = tid >> 5;
  int coT   = (f & 3) * 16;
  int chunk = f >> 2;
  int m     = lane & 15;
#pragma unroll
  for (int i = 0; i < 16; ++i)
    dst[(size_t)(f * 32 + lane) * 16 + i] =
        (_Float16)w[(coT + m) * K + chunk * 32 + a_kidx(lane, i)];
}

// Conv weights w[((co*64+ci)*27)+t] -> dst[frag][lane][i], frag = (t*2+chunk)*4 + coTile
__global__ void k_pack_a_conv(const float* __restrict__ w, _Float16* __restrict__ dst) {
  int tid = blockIdx.x * blockDim.x + threadIdx.x;   // 27*2*4*32 = 6912
  if (tid >= 27 * 2 * 4 * 32) return;
  int lane  = tid & 31;
  int f     = tid >> 5;
  int coT   = (f & 3) * 16;
  int chunk = (f >> 2) & 1;
  int t     = f >> 3;
  int m     = lane & 15;
#pragma unroll
  for (int i = 0; i < 16; ++i) {
    int ci = chunk * 32 + a_kidx(lane, i);
    dst[(size_t)(f * 32 + lane) * 16 + i] =
        (_Float16)w[((coT + m) * 64 + ci) * 27 + t];
  }
}

// ---- per-batch mean + max-norm for voxel normalization ----------------------
__global__ void k_stats(const float* __restrict__ coords, float* __restrict__ stats) {
  __shared__ float s0[256], s1[256], s2[256];
  int b = blockIdx.x, t = threadIdx.x;
  float sx = 0.f, sy = 0.f, sz = 0.f;
  for (int n = t; n < NPTS; n += 256) {
    sx += coords[(b * 3 + 0) * NPTS + n];
    sy += coords[(b * 3 + 1) * NPTS + n];
    sz += coords[(b * 3 + 2) * NPTS + n];
  }
  s0[t] = sx; s1[t] = sy; s2[t] = sz;
  __syncthreads();
  for (int o = 128; o > 0; o >>= 1) {
    if (t < o) { s0[t] += s0[t + o]; s1[t] += s1[t + o]; s2[t] += s2[t + o]; }
    __syncthreads();
  }
  float mx = s0[0] / NPTS, my = s1[0] / NPTS, mz = s2[0] / NPTS;
  __syncthreads();
  float mmax = 0.f;
  for (int n = t; n < NPTS; n += 256) {
    float dx = coords[(b * 3 + 0) * NPTS + n] - mx;
    float dy = coords[(b * 3 + 1) * NPTS + n] - my;
    float dz = coords[(b * 3 + 2) * NPTS + n] - mz;
    mmax = fmaxf(mmax, sqrtf(dx * dx + dy * dy + dz * dz));
  }
  s0[t] = mmax;
  __syncthreads();
  for (int o = 128; o > 0; o >>= 1) {
    if (t < o) s0[t] = fmaxf(s0[t], s0[t + o]);
    __syncthreads();
  }
  if (t == 0) {
    stats[b * 4 + 0] = mx; stats[b * 4 + 1] = my; stats[b * 4 + 2] = mz;
    stats[b * 4 + 3] = 1.0f / (2.0f * s0[0]);
  }
}

// ---- voxelize: scatter-add features + counts, emit float voxel coords -------
__global__ void k_voxelize(const float* __restrict__ feats, const float* __restrict__ coords,
                           const float* __restrict__ stats, float* __restrict__ grid,
                           float* __restrict__ cnt, float* __restrict__ nc) {
  int pid = blockIdx.x;
  int b = pid / NPTS, n = pid % NPTS, c = threadIdx.x;
  float v[3]; int vi[3];
#pragma unroll
  for (int d = 0; d < 3; ++d) {
    float x = (coords[(b * 3 + d) * NPTS + n] - stats[b * 4 + d]) * stats[b * 4 + 3] + 0.5f;
    x = fminf(fmaxf(x * (float)RG, 0.0f), (float)RG - 1.0f);
    v[d] = x;
    vi[d] = (int)rintf(x);
  }
  int flat = (vi[0] * RG + vi[1]) * RG + vi[2];
  if (c < 3) nc[(b * 3 + c) * NPTS + n] = v[c];
  if (c == 0) atomicAdd(&cnt[b * RV3 + flat], 1.0f);
  atomicAdd(&grid[(size_t)(b * COUT + c) * RV3 + flat], feats[(b * CIN + c) * NPTS + n]);
}

__global__ void k_avg(float* __restrict__ grid, const float* __restrict__ cnt) {
  long i = (long)blockIdx.x * blockDim.x + threadIdx.x;
  if (i >= (long)B_ * COUT * RV3) return;
  int vox = (int)(i % RV3);
  int b = (int)(i / ((long)COUT * RV3));
  grid[i] /= fmaxf(cnt[b * RV3 + vox], 1.0f);
}

// ---- 3D conv 3x3x3 (SAME) + bias + BN + leaky ReLU via WMMA implicit GEMM ---
// block = 128 threads (4 waves). N-tile = 16 voxels along z, M = all 64 co.
// LDS layout: [plane 0..8][zz 0..17][ci 0..63] so B fragments are contiguous.
__global__ void __launch_bounds__(128)
k_conv3d_wmma(const float* __restrict__ in, float* __restrict__ out,
              const _Float16* __restrict__ wpk, const float* __restrict__ bias,
              const float* __restrict__ gamma, const float* __restrict__ beta) {
  __shared__ _Float16 lds[9 * 18 * 64];
  int bid = blockIdx.x;
  int b  = bid >> 11;
  int r  = bid & 2047;
  int x  = r >> 6;
  int y  = (r >> 1) & 31;
  int z0 = (r & 1) * 16;
  int tid = threadIdx.x;

  // stage halo input, f32 -> f16 (read z-contiguous, write [row][ci])
  for (int e = tid; e < 9 * 64 * 18; e += 128) {
    int p   = e / (64 * 18);
    int rem = e % (64 * 18);
    int ci  = rem / 18;
    int zz  = rem % 18;
    int xx  = x + p / 3 - 1;
    int yy  = y + p % 3 - 1;
    int z   = z0 + zz - 1;
    float v = 0.0f;
    if (xx >= 0 && xx < RG && yy >= 0 && yy < RG && z >= 0 && z < RG)
      v = in[(size_t)(b * COUT + ci) * RV3 + (xx * RG + yy) * RG + z];
    lds[(p * 18 + zz) * 64 + ci] = (_Float16)v;
  }
  __syncthreads();

  int lane = tid & 31;
  int wave = tid >> 5;
  int coT  = wave * 16;
  int mrow = lane & 15;     // A row / B col / D col
  int hi   = lane >> 4;

  const v16h* apk = (const v16h*)wpk;
  v8f acc = {};
  for (int t = 0; t < 27; ++t) {
    int dx = t / 9, dy = (t / 3) % 3, dz = t % 3;
    const _Float16* rowp = &lds[((dx * 3 + dy) * 18 + (mrow + dz)) * 64];
#pragma unroll
    for (int kb = 0; kb < 64; kb += 32) {
      v16h a  = apk[(size_t)((t * 2 + (kb >> 5)) * 4 + wave) * 32 + lane];  // 32B load
      v16h bf = *(const v16h*)&rowp[kb + hi * 16];                          // 32B LDS load
      acc = __builtin_amdgcn_wmma_f32_16x16x32_f16(false, a, false, bf,
                                                   (short)0, acc, false, false);
    }
  }

  // epilogue: bias + eval BN + leaky ReLU(0.1)
  int z = z0 + mrow;
#pragma unroll
  for (int r8 = 0; r8 < 8; ++r8) {
    int co = coT + hi * 8 + r8;
    float v = acc[r8] + bias[co];
    v = v * (gamma[co] * rsqrtf(1.0f + BNEPS)) + beta[co];
    v = (v >= 0.0f) ? v : 0.1f * v;
    out[(size_t)(b * COUT + co) * RV3 + (x * RG + y) * RG + z] = v;
  }
}

// ---- trilinear devoxelize ---------------------------------------------------
__global__ void k_devox(const float* __restrict__ grid, const float* __restrict__ nc,
                        float* __restrict__ f1) {
  int pid = blockIdx.x * blockDim.x + threadIdx.x;
  if (pid >= B_ * NPTS) return;
  int b = pid / NPTS, n = pid % NPTS;
  float c0 = nc[(b * 3 + 0) * NPTS + n];
  float c1 = nc[(b * 3 + 1) * NPTS + n];
  float c2 = nc[(b * 3 + 2) * NPTS + n];
  float l0 = floorf(c0), l1 = floorf(c1), l2 = floorf(c2);
  float w0 = c0 - l0, w1 = c1 - l1, w2 = c2 - l2;
  int i0 = (int)l0, i1 = (int)l1, i2 = (int)l2;
  int h0 = min(i0 + 1, RG - 1), h1 = min(i1 + 1, RG - 1), h2 = min(i2 + 1, RG - 1);
  int fidx[8]; float w[8]; int q = 0;
#pragma unroll
  for (int dx = 0; dx < 2; ++dx)
#pragma unroll
    for (int dy = 0; dy < 2; ++dy)
#pragma unroll
      for (int dz = 0; dz < 2; ++dz) {
        int ix = dx ? h0 : i0, iy = dy ? h1 : i1, iz = dz ? h2 : i2;
        w[q]    = (dx ? w0 : 1.f - w0) * (dy ? w1 : 1.f - w1) * (dz ? w2 : 1.f - w2);
        fidx[q] = (ix * RG + iy) * RG + iz;
        ++q;
      }
  for (int c = 0; c < COUT; ++c) {
    const float* g = grid + (size_t)(b * COUT + c) * RV3;
    float s = 0.f;
#pragma unroll
    for (int e = 0; e < 8; ++e) s += g[fidx[e]] * w[e];
    f1[(b * COUT + c) * NPTS + n] = s;
  }
}

// ---- 64x64 GEMM over points via WMMA: q/k/v (mode 0) and f3 (mode 1) --------
// LDS layout [n 0..15][ci 0..63] -> B fragments contiguous.
__global__ void __launch_bounds__(128)
k_linear_wmma(const float* __restrict__ X, const _Float16* __restrict__ wpk,
              const float* __restrict__ bias, const float* __restrict__ gamma,
              const float* __restrict__ beta, float* __restrict__ out, int mode) {
  __shared__ _Float16 lds[16 * 64];
  int bid = blockIdx.x;
  int b  = bid >> 8;              // NPTS/16 = 256 tiles per batch
  int n0 = (bid & 255) * 16;
  int tid = threadIdx.x;
  for (int e = tid; e < 64 * 16; e += 128) {
    int ci = e >> 4, n = e & 15;
    lds[n * 64 + ci] = (_Float16)X[(b * CIN + ci) * NPTS + n0 + n];
  }
  __syncthreads();

  int lane = tid & 31, wave = tid >> 5;
  int coT = wave * 16, mrow = lane & 15, hi = lane >> 4;
  const v16h* apk = (const v16h*)wpk;
  v8f acc = {};
#pragma unroll
  for (int kb = 0; kb < 64; kb += 32) {
    v16h a  = apk[(size_t)(((kb >> 5) * 4 + wave) * 32) + lane];
    v16h bf = *(const v16h*)&lds[mrow * 64 + kb + hi * 16];
    acc = __builtin_amdgcn_wmma_f32_16x16x32_f16(false, a, false, bf,
                                                 (short)0, acc, false, false);
  }
#pragma unroll
  for (int r8 = 0; r8 < 8; ++r8) {
    int co = coT + hi * 8 + r8;
    float v = acc[r8] + bias[co];
    if (mode == 1) {
      v = v * (gamma[co] * rsqrtf(1.0f + BNEPS)) + beta[co];
      v = fmaxf(v, 0.0f);
    }
    out[(b * COUT + co) * NPTS + n0 + mrow] = v;
  }
}

// ---- brute-force kNN (smallest K squared distances) -------------------------
__global__ void k_knn(const float* __restrict__ coords, int* __restrict__ knn_idx,
                      float* __restrict__ knn_d2) {
  __shared__ float sx[256], sy[256], sz[256];
  int blk = blockIdx.x;
  int b = blk >> 4;
  int n = (blk & 15) * 256 + threadIdx.x;
  float px = coords[(b * 3 + 0) * NPTS + n];
  float py = coords[(b * 3 + 1) * NPTS + n];
  float pz = coords[(b * 3 + 2) * NPTS + n];
  float bd[KNN]; int bi[KNN];
#pragma unroll
  for (int k = 0; k < KNN; ++k) { bd[k] = 3.4e38f; bi[k] = 0; }
  for (int m0 = 0; m0 < NPTS; m0 += 256) {
    int t = threadIdx.x;
    sx[t] = coords[(b * 3 + 0) * NPTS + m0 + t];
    sy[t] = coords[(b * 3 + 1) * NPTS + m0 + t];
    sz[t] = coords[(b * 3 + 2) * NPTS + m0 + t];
    if (m0 + 256 < NPTS)   // hint next tile into cache (global_prefetch_b8)
      __builtin_prefetch(&coords[(b * 3 + 0) * NPTS + m0 + 256 + t], 0, 1);
    __syncthreads();
    for (int j = 0; j < 256; ++j) {
      float dx = px - sx[j], dy = py - sy[j], dz = pz - sz[j];
      float d2 = dx * dx + dy * dy + dz * dz;
      if (d2 < bd[KNN - 1]) {
        int q = KNN - 1;
        while (q > 0 && bd[q - 1] > d2) { bd[q] = bd[q - 1]; bi[q] = bi[q - 1]; --q; }
        bd[q] = d2; bi[q] = m0 + j;
      }
    }
    __syncthreads();
  }
  for (int k = 0; k < KNN; ++k) {
    knn_idx[(b * NPTS + n) * KNN + k] = bi[k];
    knn_d2 [(b * NPTS + n) * KNN + k] = bd[k];
  }
}

// ---- radius-masked single-head attention over K neighbors -------------------
__global__ void k_attn(const float* __restrict__ q, const float* __restrict__ kf,
                       const float* __restrict__ vf, const int* __restrict__ knn_idx,
                       const float* __restrict__ knn_d2, float* __restrict__ f2) {
  int pid = blockIdx.x * blockDim.x + threadIdx.x;
  if (pid >= B_ * NPTS) return;
  int b = pid / NPTS, n = pid % NPTS;
  float sc[KNN];
  float smax = -3.4e38f;
  for (int k = 0; k < KNN; ++k) {
    int m = knn_idx[(b * NPTS + n) * KNN + k];
    float s = 0.f;
    for (int c = 0; c < COUT; ++c)
      s += q[(b * COUT + c) * NPTS + n] * kf[(b * COUT + c) * NPTS + m];
    s *= 0.125f;                                  // 1/sqrt(64)
    if (knn_d2[(b * NPTS + n) * KNN + k] > RAD2) s = -1e9f;
    sc[k] = s;
    smax = fmaxf(smax, s);
  }
  float ssum = 0.f;
#pragma unroll
  for (int k = 0; k < KNN; ++k) { sc[k] = __expf(sc[k] - smax); ssum += sc[k]; }
  float inv = 1.0f / ssum;
  for (int c = 0; c < COUT; ++c) {
    float o = 0.f;
#pragma unroll
    for (int k = 0; k < KNN; ++k) {
      int m = knn_idx[(b * NPTS + n) * KNN + k];
      o += sc[k] * vf[(b * COUT + c) * NPTS + m];
    }
    f2[(b * COUT + c) * NPTS + n] = o * inv;
  }
}

// ---- final fusion: w2 x concat(f1,f2) + b2 + f3 -> d_out (WMMA, K=128) ------
__global__ void __launch_bounds__(128)
k_fuse_wmma(const float* __restrict__ f1, const float* __restrict__ f2,
            const _Float16* __restrict__ w2pk, const float* __restrict__ b2,
            const float* __restrict__ f3, float* __restrict__ out) {
  __shared__ _Float16 lds[16 * 128];
  int bid = blockIdx.x;
  int b  = bid >> 8;
  int n0 = (bid & 255) * 16;
  int tid = threadIdx.x;
  for (int e = tid; e < 128 * 16; e += 128) {
    int ci = e >> 4, n = e & 15;
    float v = (ci < 64) ? f1[(b * COUT + ci) * NPTS + n0 + n]
                        : f2[(b * COUT + (ci - 64)) * NPTS + n0 + n];
    lds[n * 128 + ci] = (_Float16)v;
  }
  __syncthreads();

  int lane = tid & 31, wave = tid >> 5;
  int coT = wave * 16, mrow = lane & 15, hi = lane >> 4;
  const v16h* apk = (const v16h*)w2pk;
  v8f acc = {};
#pragma unroll
  for (int kb = 0; kb < 128; kb += 32) {
    v16h a  = apk[(size_t)(((kb >> 5) * 4 + wave) * 32) + lane];
    v16h bf = *(const v16h*)&lds[mrow * 128 + kb + hi * 16];
    acc = __builtin_amdgcn_wmma_f32_16x16x32_f16(false, a, false, bf,
                                                 (short)0, acc, false, false);
  }
#pragma unroll
  for (int r8 = 0; r8 < 8; ++r8) {
    int co = coT + hi * 8 + r8;
    int n  = n0 + mrow;
    out[(b * COUT + co) * NPTS + n] = acc[r8] + b2[co] + f3[(b * COUT + co) * NPTS + n];
  }
}

// ============================================================================
extern "C" void kernel_launch(void* const* d_in, const int* in_sizes, int n_in,
                              void* d_out, int out_size, void* d_ws, size_t ws_size,
                              hipStream_t stream) {
  (void)in_sizes; (void)n_in; (void)out_size; (void)ws_size;
  const float* features = (const float*)d_in[0];
  const float* coords   = (const float*)d_in[1];
  const float* w_v1 = (const float*)d_in[2];
  const float* b_v1 = (const float*)d_in[3];
  const float* g1   = (const float*)d_in[4];
  const float* be1  = (const float*)d_in[5];
  const float* w_v2 = (const float*)d_in[6];
  const float* b_v2 = (const float*)d_in[7];
  const float* g2   = (const float*)d_in[8];
  const float* be2  = (const float*)d_in[9];
  const float* wq = (const float*)d_in[10]; const float* bq  = (const float*)d_in[11];
  const float* wk = (const float*)d_in[12]; const float* bk  = (const float*)d_in[13];
  const float* wv = (const float*)d_in[14]; const float* bv  = (const float*)d_in[15];
  const float* wp = (const float*)d_in[16]; const float* bp  = (const float*)d_in[17];
  const float* gp = (const float*)d_in[18]; const float* bep = (const float*)d_in[19];
  const float* w2 = (const float*)d_in[20]; const float* b2  = (const float*)d_in[21];
  float* outp = (float*)d_out;

  // ---- workspace partition (256B aligned) ----
  char* ws = (char*)d_ws;
  size_t off = 0;
  auto alloc = [&](size_t bytes) -> void* {
    void* p = ws + off;
    off += (bytes + 255) & ~(size_t)255;
    return p;
  };
  float* stats   = (float*)alloc((size_t)B_ * 4 * sizeof(float));
  float* cnt     = (float*)alloc((size_t)B_ * RV3 * sizeof(float));
  float* ncb     = (float*)alloc((size_t)B_ * 3 * NPTS * sizeof(float));
  float* gridA   = (float*)alloc((size_t)B_ * COUT * RV3 * sizeof(float));
  float* gridB   = (float*)alloc((size_t)B_ * COUT * RV3 * sizeof(float));
  float* f1      = (float*)alloc((size_t)B_ * COUT * NPTS * sizeof(float));
  float* f2      = (float*)alloc((size_t)B_ * COUT * NPTS * sizeof(float));
  float* qbuf    = (float*)alloc((size_t)B_ * COUT * NPTS * sizeof(float));
  float* kbuf    = (float*)alloc((size_t)B_ * COUT * NPTS * sizeof(float));
  float* vbuf    = (float*)alloc((size_t)B_ * COUT * NPTS * sizeof(float));
  float* f3      = (float*)alloc((size_t)B_ * COUT * NPTS * sizeof(float));
  int*   knn_idx = (int*)  alloc((size_t)B_ * NPTS * KNN * sizeof(int));
  float* knn_d2  = (float*)alloc((size_t)B_ * NPTS * KNN * sizeof(float));
  _Float16* wv1h = (_Float16*)alloc((size_t)64 * 64 * 27 * sizeof(_Float16));
  _Float16* wv2h = (_Float16*)alloc((size_t)64 * 64 * 27 * sizeof(_Float16));
  _Float16* wqh  = (_Float16*)alloc((size_t)64 * 64 * sizeof(_Float16));
  _Float16* wkh  = (_Float16*)alloc((size_t)64 * 64 * sizeof(_Float16));
  _Float16* wvh  = (_Float16*)alloc((size_t)64 * 64 * sizeof(_Float16));
  _Float16* wph  = (_Float16*)alloc((size_t)64 * 64 * sizeof(_Float16));
  _Float16* w2h  = (_Float16*)alloc((size_t)64 * 128 * sizeof(_Float16));

  // weight repack into WMMA fragment order (A fragments as contiguous 32B)
  k_pack_a_conv<<<(27 * 2 * 4 * 32 + 255) / 256, 256, 0, stream>>>(w_v1, wv1h);
  k_pack_a_conv<<<(27 * 2 * 4 * 32 + 255) / 256, 256, 0, stream>>>(w_v2, wv2h);
  k_pack_a_gemm<<<1, 256, 0, stream>>>(wq, wqh, 64);
  k_pack_a_gemm<<<1, 256, 0, stream>>>(wk, wkh, 64);
  k_pack_a_gemm<<<1, 256, 0, stream>>>(wv, wvh, 64);
  k_pack_a_gemm<<<1, 256, 0, stream>>>(wp, wph, 64);
  k_pack_a_gemm<<<2, 256, 0, stream>>>(w2, w2h, 128);

  // voxelization
  k_stats<<<B_, 256, 0, stream>>>(coords, stats);
  {
    long nz = (long)B_ * COUT * RV3;
    k_zero<<<(int)((nz + 255) / 256), 256, 0, stream>>>(gridA, nz);
    long nc2 = (long)B_ * RV3;
    k_zero<<<(int)((nc2 + 255) / 256), 256, 0, stream>>>(cnt, nc2);
  }
  k_voxelize<<<B_ * NPTS, 64, 0, stream>>>(features, coords, stats, gridA, cnt, ncb);
  {
    long nz = (long)B_ * COUT * RV3;
    k_avg<<<(int)((nz + 255) / 256), 256, 0, stream>>>(gridA, cnt);
  }

  // two conv+BN+leaky stages (grid ping-pong, L2-resident)
  k_conv3d_wmma<<<B_ * RG * RG * 2, 128, 0, stream>>>(gridA, gridB, wv1h, b_v1, g1, be1);
  k_conv3d_wmma<<<B_ * RG * RG * 2, 128, 0, stream>>>(gridB, gridA, wv2h, b_v2, g2, be2);

  // devoxelize -> f1
  k_devox<<<(B_ * NPTS + 63) / 64, 64, 0, stream>>>(gridA, ncb, f1);

  // point branch projections (WMMA GEMMs) + f3
  int gblk = B_ * (NPTS / 16);
  k_linear_wmma<<<gblk, 128, 0, stream>>>(features, wqh, bq, nullptr, nullptr, qbuf, 0);
  k_linear_wmma<<<gblk, 128, 0, stream>>>(features, wkh, bk, nullptr, nullptr, kbuf, 0);
  k_linear_wmma<<<gblk, 128, 0, stream>>>(features, wvh, bv, nullptr, nullptr, vbuf, 0);
  k_linear_wmma<<<gblk, 128, 0, stream>>>(features, wph, bp, gp, bep, f3, 1);

  // kNN + attention -> f2
  k_knn<<<B_ * (NPTS / 256), 256, 0, stream>>>(coords, knn_idx, knn_d2);
  k_attn<<<(B_ * NPTS + 63) / 64, 64, 0, stream>>>(qbuf, kbuf, vbuf, knn_idx, knn_d2, f2);

  // fusion -> d_out[0 : B*C*N], then coords -> d_out tail
  k_fuse_wmma<<<gblk, 128, 0, stream>>>(f1, f2, w2h, b2, f3, outp);
  k_copy<<<(B_ * 3 * NPTS + 255) / 256, 256, 0, stream>>>(
      coords, outp + (size_t)B_ * COUT * NPTS, B_ * 3 * NPTS);
}
